// TreeAttentionV2_52639119180126
// MI455X (gfx1250) — matrix operand order repaired and checked by reading, not compiled, not run on hardware
//
#include <hip/hip_runtime.h>
#include <hip/hip_bf16.h>

// ---------------------------------------------------------------------------
// Types & WMMA helpers (CDNA5 / gfx1250, wave32)
// ---------------------------------------------------------------------------
typedef __bf16 bf16_t;
typedef __attribute__((ext_vector_type(16))) __bf16 v16bf;
typedef __attribute__((ext_vector_type(8)))  float  v8f;

struct Q2 { uint4 lo, hi; };

static __device__ inline v16bf make_frag(uint4 lo, uint4 hi) {
  Q2 t; t.lo = lo; t.hi = hi;
  return __builtin_bit_cast(v16bf, t);
}

static __device__ inline v8f zero_v8f() {
  v8f z;
#pragma unroll
  for (int i = 0; i < 8; ++i) z[i] = 0.0f;
  return z;
}

static __device__ inline v8f wmma_bf16(v16bf a, v16bf b, v8f c) {
  // D = A(16x32) * B(32x16) + C(16x16), fp32 accumulate
  return __builtin_amdgcn_wmma_f32_16x16x32_bf16(
      /*neg_a=*/false, a, /*neg_b=*/false, b,
      /*c_mod=*/(short)0, c, /*reuse_a=*/false, /*reuse_b=*/false);
}

// ---------------------------------------------------------------------------
// Generic GEMM:  C[M,N] = (A[M,K] * B[K,N] + bias[N]) * colscale
// colscale = `scale` for col < scale_cols else 1 (folds 1/sqrt(hd) into Q).
// Block = 256 threads (8 waves), tile 128x128, K-step 32, register-prefetch
// software pipeline so global loads of tile k+1 overlap WMMAs of tile k.
// ---------------------------------------------------------------------------
template <typename AT, typename OT>
__global__ __launch_bounds__(256)
void gemm_bias_kernel(const AT* __restrict__ A, const float* __restrict__ B,
                      const float* __restrict__ bias, OT* __restrict__ Cout,
                      int M, int N, int K, int scale_cols, float scale) {
  constexpr int LS = 40;  // padded row stride (bf16 elems); 80B keeps 16B align
  __shared__ bf16_t sA[128 * LS];  // [m][k]
  __shared__ bf16_t sB[128 * LS];  // [n][k]  (transposed stage)

  const int tid  = threadIdx.x;
  const int lane = tid & 31;
  const int wid  = tid >> 5;
  const int lh   = lane >> 4;   // half-wave (0/1)
  const int ln   = lane & 15;

  const int mbase = blockIdx.y * 128;
  const int nbase = blockIdx.x * 128;
  const int mwave = (wid >> 1) * 32;
  const int nwave = (wid & 1) * 64;

  // cooperative-stage mapping
  const int arow = tid >> 1;            // 0..127
  const int acol = (tid & 1) * 16;      // 0 / 16
  const int bn   = tid & 127;           // 0..127
  const int bk0  = (tid >> 7) * 16;     // 0 / 16

  v8f acc[2][4];
#pragma unroll
  for (int i = 0; i < 2; ++i)
#pragma unroll
    for (int j = 0; j < 4; ++j) acc[i][j] = zero_v8f();

  float areg[16], breg[16];
  // prologue fetch (kb = 0)
  {
    const AT* ga = A + (size_t)(mbase + arow) * K + acol;
#pragma unroll
    for (int j = 0; j < 16; ++j) areg[j] = (float)ga[j];
#pragma unroll
    for (int kk = 0; kk < 16; ++kk)
      breg[kk] = B[(size_t)(bk0 + kk) * N + nbase + bn];
  }

  for (int kb = 0; kb < K; kb += 32) {
    __syncthreads();
#pragma unroll
    for (int j = 0; j < 16; ++j)
      sA[arow * LS + acol + j] = (bf16_t)areg[j];
#pragma unroll
    for (int kk = 0; kk < 16; ++kk)
      sB[bn * LS + bk0 + kk] = (bf16_t)breg[kk];
    __syncthreads();

    // prefetch next tile into registers while this tile's WMMAs run
    if (kb + 32 < K) {
      const AT* ga = A + (size_t)(mbase + arow) * K + (kb + 32) + acol;
#pragma unroll
      for (int j = 0; j < 16; ++j) areg[j] = (float)ga[j];
#pragma unroll
      for (int kk = 0; kk < 16; ++kk)
        breg[kk] = B[(size_t)(kb + 32 + bk0 + kk) * N + nbase + bn];
    }

    // Build fragments in contiguous k order (A and B match).
    // A (16x32 bf16): lane ln = row; elems e -> K = lh*16+e (16 contiguous)
    v16bf afrag[2];
#pragma unroll
    for (int i = 0; i < 2; ++i) {
      const bf16_t* p = &sA[(mwave + i * 16 + ln) * LS];
      uint4 lo = *(const uint4*)(p + lh * 16);
      uint4 hi = *(const uint4*)(p + lh * 16 + 8);
      afrag[i] = make_frag(lo, hi);
    }
    // B (32x16 bf16): lane ln = col; elems e -> K = lh*16+e (16 contiguous)
    v16bf bfrag[4];
#pragma unroll
    for (int j = 0; j < 4; ++j) {
      const bf16_t* p = &sB[(nwave + j * 16 + ln) * LS + lh * 16];
      bfrag[j] = make_frag(*(const uint4*)p, *(const uint4*)(p + 8));
    }
#pragma unroll
    for (int i = 0; i < 2; ++i)
#pragma unroll
      for (int j = 0; j < 4; ++j)
        acc[i][j] = wmma_bf16(afrag[i], bfrag[j], acc[i][j]);
  }

  // Epilogue: D layout -> lane ln = col, VGPR r = row (r + 8*lh)
#pragma unroll
  for (int j = 0; j < 4; ++j) {
    const int col = nbase + nwave + j * 16 + ln;
    const float bv = bias[col];
    const float sc = (col < scale_cols) ? scale : 1.0f;
#pragma unroll
    for (int i = 0; i < 2; ++i) {
#pragma unroll
      for (int r = 0; r < 8; ++r) {
        const int row = mbase + mwave + i * 16 + lh * 8 + r;
        Cout[(size_t)row * N + col] = (OT)((acc[i][j][r] + bv) * sc);
      }
    }
  }
}

// ---------------------------------------------------------------------------
// Fused attention per (b, h): whole K head [1024x64] + V^T head [64x1024]
// resident in LDS as bf16 (CDNA5 320KB WGP LDS). K head is DMA'd with
// global_load_async_to_lds_b128 (ASYNCcnt path); V is transposed manually.
// Q is pre-scaled by 1/sqrt(hd) in the QKV GEMM epilogue, so scores come out
// of the WMMAs already scaled. 8 waves; each wave owns 16-row Q tiles.
// 3-pass flash: (max) -> (sum of exp) -> (threshold + P@V).
// relu(softmax - 0.001) applied post-normalization (needs Z before PV).
// ---------------------------------------------------------------------------
__global__ __launch_bounds__(256)
void attn_kernel(const bf16_t* __restrict__ qkv, bf16_t* __restrict__ ybuf) {
  constexpr int T = 1024, HD = 64, C = 1024, C3 = 3072;
  constexpr int VP = 1040;  // padded V^T row stride (keeps 16B align: 2080B)
  extern __shared__ bf16_t smem[];
  bf16_t* sK = smem;                 // [T][HD]   key-major (score B frags)
  bf16_t* sV = sK + T * HD;          // [HD][VP]  V transposed (PV B frags)
  bf16_t* sP = sV + HD * VP;         // 8 waves x [16][32] P staging

  const int b    = blockIdx.x >> 4;
  const int h    = blockIdx.x & 15;
  const int tid  = threadIdx.x;
  const int lane = tid & 31;
  const int wid  = tid >> 5;
  const int lh   = lane >> 4;
  const int ln   = lane & 15;

  // ---- stage K via async DMA to LDS: 8192 x 16B chunks, LDS-contiguous ----
  {
    const uint64_t gbase = (uint64_t)(uintptr_t)qkv +
        ((uint64_t)((size_t)b * T) * C3 + C + (uint64_t)h * HD) * sizeof(bf16_t);
    const uint32_t lbase = (uint32_t)(uintptr_t)sK;  // LDS aperture: low 32b = offset
    for (int chunk = tid; chunk < (T * HD) / 8; chunk += 256) {
      const int t = chunk >> 3;       // key row
      const int g = chunk & 7;        // 16B group within 128B row
      const uint64_t ga = gbase + (uint64_t)t * (C3 * sizeof(bf16_t)) + g * 16;
      const uint32_t la = lbase + (uint32_t)chunk * 16;
      asm volatile("global_load_async_to_lds_b128 %0, %1, off"
                   :: "v"(la), "v"(ga) : "memory");
    }
  }
  // ---- stage V^T (transpose; regular ds stores) ----
  for (int idx = tid; idx < T * HD; idx += 256) {
    const int t = idx >> 6;
    const int d = idx & 63;
    sV[d * VP + t] =
        qkv[(size_t)(b * T + t) * C3 + 2 * C + (size_t)h * HD + d];
  }
  asm volatile("s_wait_asynccnt 0x0" ::: "memory");
  __syncthreads();

  bf16_t* myP = sP + wid * (16 * 32);

  for (int qt = wid; qt < T / 16; qt += 8) {
    const int q0 = qt * 16;

    // Q fragments (16x64 = two 16x32 A fragments), direct from global bf16.
    v16bf aq[2];
    const size_t qrow = (size_t)(b * T + q0 + ln) * C3 + (size_t)h * HD;
#pragma unroll
    for (int c = 0; c < 2; ++c) {
      uint4 lo = *(const uint4*)&qkv[qrow + c * 32 + lh * 16];
      uint4 hi = *(const uint4*)&qkv[qrow + c * 32 + lh * 16 + 8];
      aq[c] = make_frag(lo, hi);
    }

    // ---- pass A: row max (lane-local, reduce once at end) ----
    float lmax[8];
#pragma unroll
    for (int r = 0; r < 8; ++r) lmax[r] = -1e30f;
    for (int kb = 0; kb < T / 16; ++kb) {
      const int key = kb * 16 + ln;
      v16bf bk[2];
#pragma unroll
      for (int c = 0; c < 2; ++c) {
        const bf16_t* p = &sK[key * HD + c * 32 + lh * 16];
        bk[c] = make_frag(*(const uint4*)p, *(const uint4*)(p + 8));
      }
      v8f s = wmma_bf16(aq[0], bk[0], zero_v8f());
      s = wmma_bf16(aq[1], bk[1], s);
#pragma unroll
      for (int r = 0; r < 8; ++r) lmax[r] = fmaxf(lmax[r], s[r]);
    }
    float rmax[8];
#pragma unroll
    for (int r = 0; r < 8; ++r) {  // reduce over 16 lanes of this half-wave
      float tm = lmax[r];
      tm = fmaxf(tm, __shfl_xor(tm, 1));
      tm = fmaxf(tm, __shfl_xor(tm, 2));
      tm = fmaxf(tm, __shfl_xor(tm, 4));
      tm = fmaxf(tm, __shfl_xor(tm, 8));
      rmax[r] = tm;
    }

    // ---- pass B: sum of exp ----
    float lsum[8];
#pragma unroll
    for (int r = 0; r < 8; ++r) lsum[r] = 0.0f;
    for (int kb = 0; kb < T / 16; ++kb) {
      const int key = kb * 16 + ln;
      v16bf bk[2];
#pragma unroll
      for (int c = 0; c < 2; ++c) {
        const bf16_t* p = &sK[key * HD + c * 32 + lh * 16];
        bk[c] = make_frag(*(const uint4*)p, *(const uint4*)(p + 8));
      }
      v8f s = wmma_bf16(aq[0], bk[0], zero_v8f());
      s = wmma_bf16(aq[1], bk[1], s);
#pragma unroll
      for (int r = 0; r < 8; ++r) lsum[r] += __expf(s[r] - rmax[r]);
    }
    float rinv[8];
#pragma unroll
    for (int r = 0; r < 8; ++r) {
      float e = lsum[r];
      e += __shfl_xor(e, 1);
      e += __shfl_xor(e, 2);
      e += __shfl_xor(e, 4);
      e += __shfl_xor(e, 8);
      rinv[r] = 1.0f / e;
    }

    // ---- pass C: P = relu(exp(s-m)/Z - 0.001); O += P @ V ----
    v8f oacc[4];
#pragma unroll
    for (int n = 0; n < 4; ++n) oacc[n] = zero_v8f();

    for (int ks = 0; ks < T / 32; ++ks) {
#pragma unroll
      for (int t2 = 0; t2 < 2; ++t2) {
        const int key = ks * 32 + t2 * 16 + ln;
        v16bf bk[2];
#pragma unroll
        for (int c = 0; c < 2; ++c) {
          const bf16_t* p = &sK[key * HD + c * 32 + lh * 16];
          bk[c] = make_frag(*(const uint4*)p, *(const uint4*)(p + 8));
        }
        v8f s = wmma_bf16(aq[0], bk[0], zero_v8f());
        s = wmma_bf16(aq[1], bk[1], s);
#pragma unroll
        for (int r = 0; r < 8; ++r) {
          float p = __expf(s[r] - rmax[r]) * rinv[r] - 0.001f;
          p = fmaxf(p, 0.0f);
          // D layout (lane=col, row = lh*8+r) -> row-major P staging
          myP[(lh * 8 + r) * 32 + t2 * 16 + ln] = (bf16_t)p;
        }
      }
      // wave-private LDS RAW: DS ops are in-order within a wave; make sure
      // the compiler doesn't hoist the loads.
      asm volatile("s_wait_dscnt 0" ::: "memory");

      // Reload P as an A fragment (16 rows x 32 keys).
      uint4 plo = *(const uint4*)&myP[ln * 32 + lh * 16];
      uint4 phi = *(const uint4*)&myP[ln * 32 + lh * 16 + 8];
      v16bf ap = make_frag(plo, phi);

#pragma unroll
      for (int n = 0; n < 4; ++n) {
        const bf16_t* p = &sV[(n * 16 + ln) * VP + ks * 32 + lh * 16];
        v16bf bv = make_frag(*(const uint4*)p, *(const uint4*)(p + 8));
        oacc[n] = wmma_bf16(ap, bv, oacc[n]);
      }
    }

    // Store O tile -> ybuf[b*T+t][h*64 + col] (bf16, feeds proj GEMM)
#pragma unroll
    for (int n = 0; n < 4; ++n) {
      const int col = h * HD + n * 16 + ln;
#pragma unroll
      for (int r = 0; r < 8; ++r) {
        const int t = q0 + lh * 8 + r;
        ybuf[(size_t)(b * T + t) * C + col] = (bf16_t)oacc[n][r];
      }
    }
  }
}

// ---------------------------------------------------------------------------
// Launch: qkv GEMM -> fused attention -> proj GEMM
// Workspace: qkv bf16 (8192x3072 = 48MB) + ybuf bf16 (8192x1024 = 16MB)
// ---------------------------------------------------------------------------
extern "C" void kernel_launch(void* const* d_in, const int* in_sizes, int n_in,
                              void* d_out, int out_size, void* d_ws, size_t ws_size,
                              hipStream_t stream) {
  (void)in_sizes; (void)n_in; (void)out_size; (void)ws_size;
  const float* x      = (const float*)d_in[0];
  const float* w_attn = (const float*)d_in[1];
  const float* b_attn = (const float*)d_in[2];
  const float* w_proj = (const float*)d_in[3];
  const float* b_proj = (const float*)d_in[4];
  float* out = (float*)d_out;

  const int B = 8, T = 1024, C = 1024;
  const int M = B * T;  // 8192

  bf16_t* qkv  = (bf16_t*)d_ws;              // M x 3C bf16
  bf16_t* ybuf = qkv + (size_t)M * 3 * C;    // M x C  bf16

  // 1) QKV projection: (8192 x 1024) @ (1024 x 3072) + b, Q cols pre-scaled
  //    by 1/sqrt(hd) = 0.125 so attention scores need no per-element scale.
  dim3 g1(3 * C / 128, M / 128);
  hipLaunchKernelGGL((gemm_bias_kernel<float, bf16_t>), g1, dim3(256), 0, stream,
                     x, w_attn, b_attn, qkv, M, 3 * C, C,
                     /*scale_cols=*/C, /*scale=*/0.125f);

  // 2) fused attention (one workgroup per (b,h))
  const size_t shmem =
      (size_t)(1024 * 64 + 64 * 1040 + 8 * 16 * 32) * sizeof(bf16_t);  // 272KB
  (void)hipFuncSetAttribute(reinterpret_cast<const void*>(attn_kernel),
                            hipFuncAttributeMaxDynamicSharedMemorySize,
                            (int)shmem);
  hipLaunchKernelGGL(attn_kernel, dim3(B * 16), dim3(256), shmem, stream,
                     qkv, ybuf);

  // 3) output projection: (8192 x 1024) @ (1024 x 1024) + b -> fp32 out
  dim3 g3(C / 128, M / 128);
  hipLaunchKernelGGL((gemm_bias_kernel<bf16_t, float>), g3, dim3(256), 0, stream,
                     ybuf, w_proj, b_proj, out, M, C, C,
                     /*scale_cols=*/0, /*scale=*/1.0f);
}
